// causalAttention_v1_76278619177477
// MI455X (gfx1250) — matrix operand order repaired and checked by reading, not compiled, not run on hardware
//
#include <hip/hip_runtime.h>
#include <hip/hip_bf16.h>

// Causal attention for MI455X (gfx1250, wave32, WMMA bf16 + TDM async-tensor path).
// SEQ=4096, D=1024.  O = causal_softmax((x Wq^T)(x Wk^T)^T / 32) @ (x Wv^T)
// (reference's softmax -> mask -> renorm == causal softmax exactly)
//
// d_ws layout, assumes ws_size >= ~71MB:
//   xb   : SEQ*D   bf16  ( 8 MB)
//   Wb[3]: 3*D*D   bf16  ( 6 MB)
//   Qb   : SEQ*D   bf16  ( 8 MB)  row-major
//   Kb   : SEQ*D   bf16  ( 8 MB)  row-major
//   Vt   : D*SEQ   bf16  ( 8 MB)  transposed: Vt[n][m]
//   rowM : SEQ     f32
//   rowL : SEQ     f32
//   P    : SEQ*SEQ bf16  (32 MB)  normalized probs (causal tiles + one zero tile)

#define SEQn 4096
#define DDIM 1024
#define BK   32
#define SM_SCALE 0.03125f   // 1/sqrt(1024)

typedef unsigned int u32;
typedef __attribute__((ext_vector_type(16))) __bf16 v16bf;
typedef __attribute__((ext_vector_type(8)))  float  v8f;

union Frag  { v16bf v; uint4 u[2]; };
union Pack8 { unsigned short s[8]; uint4 u; };

#if defined(__has_builtin)
#  if __has_builtin(__builtin_amdgcn_tensor_load_to_lds)
#    define HAVE_TDM 1
#  endif
#endif
#ifndef HAVE_TDM
#  define HAVE_TDM 0
#endif

__device__ __forceinline__ unsigned short f2bf(float x) {
  unsigned int u = __float_as_uint(x);
  u += 0x7FFFu + ((u >> 16) & 1u);          // round-to-nearest-even
  return (unsigned short)(u >> 16);
}

__device__ __forceinline__ v8f wmma_bf16(v16bf a, v16bf b, v8f c) {
  return __builtin_amdgcn_wmma_f32_16x16x32_bf16(false, a, false, b, (short)0, c,
                                                 false, false);
}

// ---- fragment loads (ISA 05_wmma layouts) ----
// A 16x32 bf16: lanes 0-15 K={0..7,16..23}, lanes 16-31 K={8..15,24..31}
__device__ __forceinline__ Frag fragA(const unsigned short* rowPtr, int half) {
  Frag f;
  const unsigned short* p = rowPtr + 8 * half;
  f.u[0] = *(const uint4*)p;
  f.u[1] = *(const uint4*)(p + 16);
  return f;
}
// B 32x16 bf16 from row-major B^T rows: lanes 0-15 K={0..7,16..23}, lanes 16-31 K={8..15,24..31}
__device__ __forceinline__ Frag fragB(const unsigned short* rowPtr, int half) {
  Frag f;
  const unsigned short* p = rowPtr + 8 * half;
  f.u[0] = *(const uint4*)p;
  f.u[1] = *(const uint4*)(p + 16);
  return f;
}

// ---- Tensor Data Mover: 2-D tile (rows x 32 bf16) global -> LDS ----
#if HAVE_TDM
typedef __attribute__((ext_vector_type(4))) unsigned int v4u;
typedef __attribute__((ext_vector_type(8))) int v8i;
typedef __attribute__((ext_vector_type(4))) int v4i;

__device__ __forceinline__ void tdm_issue(const void* g, u32 lds_off,
                                          u32 rows, u32 ld_elems) {
  unsigned long long ga = (unsigned long long)(uintptr_t)g;
  v4u g0;
  g0[0] = 1u;                                        // count=1, user D#
  g0[1] = lds_off;                                   // lds_addr [63:32]
  g0[2] = (u32)ga;                                   // global_addr[31:0]
  g0[3] = (u32)((ga >> 32) & 0x01FFFFFFu) | (2u << 30); // addr[56:32] | type=2
  v8i g1;
  g1[0] = (int)(1u << 16);                           // data_size=2B; mask=0
  g1[1] = (int)((ld_elems & 0xFFFFu) << 16);         // tensor_dim0 lo16
  g1[2] = (int)(((ld_elems >> 16) & 0xFFFFu) | ((rows & 0xFFFFu) << 16));
  g1[3] = (int)(((rows >> 16) & 0xFFFFu) | ((u32)BK << 16)); // tile_dim0=32
  g1[4] = (int)(rows & 0xFFFFu);                     // tile_dim1 ; tile_dim2=0
  g1[5] = (int)ld_elems;                             // tensor_dim0_stride lo32
  g1[6] = 0;                                         // stride hi / dim1_stride lo
  g1[7] = 0;
  v4i gz = {0, 0, 0, 0};
#if __clang_major__ >= 23
  v8i gz8 = {0, 0, 0, 0, 0, 0, 0, 0};
  __builtin_amdgcn_tensor_load_to_lds(g0, g1, gz, gz, gz8, 0);
#else
  __builtin_amdgcn_tensor_load_to_lds(g0, g1, gz, gz, 0);
#endif
}
__device__ __forceinline__ u32 lds_off(const void* p) {
  return (u32)(uintptr_t)p;   // low 32 bits of LDS aperture address = LDS offset
}
#else
// fallback: cooperative global->LDS copy (rows x 32 bf16, contiguous 64B rows)
__device__ __forceinline__ void coop_copy(unsigned short* dst,
                                          const unsigned short* src,
                                          int rows, int ld, int tid, int nthr) {
  int total = rows * 4;                       // 4 x uint4 per 64B row
  for (int i = tid; i < total; i += nthr) {
    int r = i >> 2, c = i & 3;
    ((uint4*)(dst + r * BK))[c] = ((const uint4*)(src + (size_t)r * ld))[c];
  }
}
#endif

// ---------------- kernel 1: f32 -> bf16 ----------------
__global__ void k_cvt_bf16(const float* __restrict__ src,
                           unsigned short* __restrict__ dst, int n) {
  int i = (blockIdx.x * blockDim.x + threadIdx.x) * 4;
  if (i >= n) return;
  float4 f = *(const float4*)(src + i);
  uint2 o;
  o.x = (unsigned int)f2bf(f.x) | ((unsigned int)f2bf(f.y) << 16);
  o.y = (unsigned int)f2bf(f.z) | ((unsigned int)f2bf(f.w) << 16);
  *(uint2*)(dst + i) = o;
}

// ---------------- kernel 2: QKV projections ----------------
// 8 waves (2m x 4n), block tile 128x256, wave tile 64x64 (16 accumulators).
// A/B tiles double-buffered in LDS, filled by the TDM.
__global__ void __launch_bounds__(256)
k_qkv_gemm(const unsigned short* __restrict__ xb,
           const unsigned short* __restrict__ Wb,
           unsigned short* __restrict__ Qb,
           unsigned short* __restrict__ Kb,
           unsigned short* __restrict__ Vtb) {
  __shared__ __align__(16) unsigned short lA[2][128 * BK];   // 2 x 8 KB
  __shared__ __align__(16) unsigned short lB[2][256 * BK];   // 2 x 16 KB
  const int z = blockIdx.z;
  const unsigned short* W = Wb + (size_t)z * DDIM * DDIM;
  const int tid = threadIdx.x, wave = tid >> 5, lane = tid & 31;
  const int half = lane >> 4, ln = lane & 15;
  const int wm = wave >> 2, wn = wave & 3;
  const int mb = blockIdx.y * 128, nb = blockIdx.x * 256;

  v8f acc[4][4];
  v8f zv = {};
  #pragma unroll
  for (int a = 0; a < 4; ++a)
    #pragma unroll
    for (int b = 0; b < 4; ++b) acc[a][b] = zv;

  const int S = DDIM / BK;   // 32 k-steps

#if HAVE_TDM
  if (wave == 0) {
    tdm_issue(xb + (size_t)mb * DDIM, lds_off(&lA[0][0]), 128, DDIM);
    tdm_issue(W  + (size_t)nb * DDIM, lds_off(&lB[0][0]), 256, DDIM);
  }
#else
  coop_copy(lA[0], xb + (size_t)mb * DDIM, 128, DDIM, tid, 256);
  coop_copy(lB[0], W  + (size_t)nb * DDIM, 256, DDIM, tid, 256);
#endif

  for (int i = 0; i < S; ++i) {
    const int p = i & 1;
#if HAVE_TDM
    if (wave == 0) {
      if (i + 1 < S) {
        const int k1 = (i + 1) * BK;
        tdm_issue(xb + (size_t)mb * DDIM + k1, lds_off(&lA[p ^ 1][0]), 128, DDIM);
        tdm_issue(W  + (size_t)nb * DDIM + k1, lds_off(&lB[p ^ 1][0]), 256, DDIM);
        __builtin_amdgcn_s_wait_tensorcnt(2);   // pair i complete (in-order)
      } else {
        __builtin_amdgcn_s_wait_tensorcnt(0);
      }
    }
#else
    if (i + 1 < S) {
      const int k1 = (i + 1) * BK;
      coop_copy(lA[p ^ 1], xb + (size_t)mb * DDIM + k1, 128, DDIM, tid, 256);
      coop_copy(lB[p ^ 1], W  + (size_t)nb * DDIM + k1, 256, DDIM, tid, 256);
    }
#endif
    __syncthreads();                            // buffer p ready everywhere
    Frag af[4];
    #pragma unroll
    for (int mi = 0; mi < 4; ++mi)
      af[mi] = fragA(&lA[p][(wm * 64 + mi * 16 + ln) * BK], half);
    #pragma unroll
    for (int ni = 0; ni < 4; ++ni) {
      Frag bf = fragB(&lB[p][(wn * 64 + ni * 16 + ln) * BK], half);
      #pragma unroll
      for (int mi = 0; mi < 4; ++mi)
        acc[mi][ni] = wmma_bf16(af[mi].v, bf.v, acc[mi][ni]);
    }
    __syncthreads();                            // done reading p before overwrite
  }

  if (z < 2) {
    unsigned short* dst = z ? Kb : Qb;
    #pragma unroll
    for (int mi = 0; mi < 4; ++mi)
      #pragma unroll
      for (int ni = 0; ni < 4; ++ni)
        #pragma unroll
        for (int r = 0; r < 8; ++r) {
          int row = mb + wm * 64 + mi * 16 + 8 * half + r;
          int col = nb + wn * 64 + ni * 16 + ln;
          dst[(size_t)row * DDIM + col] = f2bf(acc[mi][ni][r]);
        }
  } else {
    // V transposed: lane's 8 C values are 8 consecutive rows of one column.
    #pragma unroll
    for (int mi = 0; mi < 4; ++mi)
      #pragma unroll
      for (int ni = 0; ni < 4; ++ni) {
        Pack8 pk;
        #pragma unroll
        for (int r = 0; r < 8; ++r) pk.s[r] = f2bf(acc[mi][ni][r]);
        int col  = nb + wn * 64 + ni * 16 + ln;
        int rowb = mb + wm * 64 + mi * 16 + 8 * half;
        *(uint4*)(Vtb + (size_t)col * SEQn + rowb) = pk.u;
      }
  }
}

__device__ __forceinline__ float hmax16(float v) {
  #pragma unroll
  for (int m = 1; m < 16; m <<= 1) v = fmaxf(v, __shfl_xor(v, m, 32));
  return v;
}
__device__ __forceinline__ float hsum16(float v) {
  #pragma unroll
  for (int m = 1; m < 16; m <<= 1) v += __shfl_xor(v, m, 32);
  return v;
}

// A-operand load straight from global (row-major, leading dim ld)
__device__ __forceinline__ Frag loadA(const unsigned short* base, int ld,
                                      int row, int k0, int half) {
  return fragA(base + (size_t)row * ld + k0, half);
}
__device__ __forceinline__ Frag loadB(const unsigned short* base, int ld,
                                      int n, int k0, int half) {
  return fragB(base + (size_t)n * ld + k0, half);
}

// ---------------- kernel 3: online softmax statistics ----------------
__global__ void k_softmax_stats(const unsigned short* __restrict__ Qb,
                                const unsigned short* __restrict__ Kb,
                                float* __restrict__ rowM,
                                float* __restrict__ rowL) {
  __shared__ float pm[4 * 16];
  __shared__ float pl[4 * 16];
  const int mt = blockIdx.x, m0 = mt * 16;
  const int tid = threadIdx.x, wave = tid >> 5, lane = tid & 31;
  const int half = lane >> 4, ln = lane & 15;

  float mrow[8], lrow[8];
  #pragma unroll
  for (int r = 0; r < 8; ++r) { mrow[r] = -1.0e30f; lrow[r] = 0.0f; }

  for (int jt = wave; jt <= mt; jt += 4) {
    const int j0 = jt * 16;
    v8f s = {};
    for (int k0 = 0; k0 < DDIM; k0 += BK) {
      Frag a = loadA(Qb, DDIM, m0 + ln, k0, half);
      Frag b = loadB(Kb, DDIM, j0 + ln, k0, half);
      s = wmma_bf16(a.v, b.v, s);
    }
    #pragma unroll
    for (int r = 0; r < 8; ++r) {
      float v = s[r] * SM_SCALE;
      int row = m0 + 8 * half + r;
      if (jt == mt && (j0 + ln) > row) v = -__builtin_inff();
      float tmax = hmax16(v);
      float newm = fmaxf(mrow[r], tmax);
      float rs = hsum16(__expf(v - newm));
      lrow[r] = lrow[r] * __expf(mrow[r] - newm) + rs;
      mrow[r] = newm;
    }
  }

  if (ln == 0) {
    #pragma unroll
    for (int r = 0; r < 8; ++r) {
      pm[wave * 16 + 8 * half + r] = mrow[r];
      pl[wave * 16 + 8 * half + r] = lrow[r];
    }
  }
  __syncthreads();
  if (tid < 16) {
    float M = -1.0e30f;
    #pragma unroll
    for (int w = 0; w < 4; ++w) M = fmaxf(M, pm[w * 16 + tid]);
    float L = 0.0f;
    #pragma unroll
    for (int w = 0; w < 4; ++w) L += pl[w * 16 + tid] * __expf(pm[w * 16 + tid] - M);
    rowM[m0 + tid] = M;
    rowL[m0 + tid] = L;
  }
}

// ---------------- kernel 4: write normalized P (bf16) ----------------
// Tile jt==mt+1 zero-filled so 16x32 A-fragments in the PV GEMM are safe.
__global__ void k_write_p(const unsigned short* __restrict__ Qb,
                          const unsigned short* __restrict__ Kb,
                          const float* __restrict__ rowM,
                          const float* __restrict__ rowL,
                          unsigned short* __restrict__ P) {
  const int jt = blockIdx.x, mt = blockIdx.y;
  if (jt > mt + 1) return;
  const int lane = threadIdx.x & 31, half = lane >> 4, ln = lane & 15;
  const int m0 = mt * 16, j0 = jt * 16;

  if (jt == mt + 1) {
    if (lane < 16) {
      uint4 zq = make_uint4(0u, 0u, 0u, 0u);
      uint4* p = (uint4*)(P + (size_t)(m0 + lane) * SEQn + j0);
      p[0] = zq; p[1] = zq;
    }
    return;
  }

  v8f s = {};
  for (int k0 = 0; k0 < DDIM; k0 += BK) {
    Frag a = loadA(Qb, DDIM, m0 + ln, k0, half);
    Frag b = loadB(Kb, DDIM, j0 + ln, k0, half);
    s = wmma_bf16(a.v, b.v, s);
  }
  #pragma unroll
  for (int r = 0; r < 8; ++r) {
    int row = m0 + 8 * half + r;
    float p;
    if (jt == mt && (j0 + ln) > row) p = 0.0f;
    else p = __expf(s[r] * SM_SCALE - rowM[row]) / rowL[row];
    P[(size_t)row * SEQn + j0 + ln] = f2bf(p);
  }
}

// ---------------- kernel 5: O = P @ V (causal, TDM-staged) ----------------
__global__ void __launch_bounds__(256)
k_pv_gemm(const unsigned short* __restrict__ P,
          const unsigned short* __restrict__ Vt,
          float* __restrict__ out) {
  __shared__ __align__(16) unsigned short lA[2][128 * BK];
  __shared__ __align__(16) unsigned short lB[2][256 * BK];
  const int tid = threadIdx.x, wave = tid >> 5, lane = tid & 31;
  const int half = lane >> 4, ln = lane & 15;
  const int wm = wave >> 2, wn = wave & 3;
  const int mb = blockIdx.y * 128, nb = blockIdx.x * 256;

  v8f acc[4][4];
  v8f zv = {};
  #pragma unroll
  for (int a = 0; a < 4; ++a)
    #pragma unroll
    for (int b = 0; b < 4; ++b) acc[a][b] = zv;

  const int S = (mb + 128) / BK;   // causal: rows up to mb+127 need j <= mb+127

#if HAVE_TDM
  if (wave == 0) {
    tdm_issue(P  + (size_t)mb * SEQn, lds_off(&lA[0][0]), 128, SEQn);
    tdm_issue(Vt + (size_t)nb * SEQn, lds_off(&lB[0][0]), 256, SEQn);
  }
#else
  coop_copy(lA[0], P  + (size_t)mb * SEQn, 128, SEQn, tid, 256);
  coop_copy(lB[0], Vt + (size_t)nb * SEQn, 256, SEQn, tid, 256);
#endif

  for (int i = 0; i < S; ++i) {
    const int p = i & 1;
    const int k0 = i * BK;
#if HAVE_TDM
    if (wave == 0) {
      if (i + 1 < S) {
        const int k1 = (i + 1) * BK;
        tdm_issue(P  + (size_t)mb * SEQn + k1, lds_off(&lA[p ^ 1][0]), 128, SEQn);
        tdm_issue(Vt + (size_t)nb * SEQn + k1, lds_off(&lB[p ^ 1][0]), 256, SEQn);
        __builtin_amdgcn_s_wait_tensorcnt(2);
      } else {
        __builtin_amdgcn_s_wait_tensorcnt(0);
      }
    }
#else
    if (i + 1 < S) {
      const int k1 = (i + 1) * BK;
      coop_copy(lA[p ^ 1], P  + (size_t)mb * SEQn + k1, 128, SEQn, tid, 256);
      coop_copy(lB[p ^ 1], Vt + (size_t)nb * SEQn + k1, 256, SEQn, tid, 256);
    }
#endif
    __syncthreads();
    // causal guards: subtile rows rm..rm+15 consume j-tiles only up to rm+16
    bool do_mi[4];
    #pragma unroll
    for (int mi = 0; mi < 4; ++mi)
      do_mi[mi] = (k0 < mb + wm * 64 + mi * 16 + 16);
    Frag af[4];
    #pragma unroll
    for (int mi = 0; mi < 4; ++mi)
      if (do_mi[mi])
        af[mi] = fragA(&lA[p][(wm * 64 + mi * 16 + ln) * BK], half);
    #pragma unroll
    for (int ni = 0; ni < 4; ++ni) {
      Frag bf = fragB(&lB[p][(wn * 64 + ni * 16 + ln) * BK], half);
      #pragma unroll
      for (int mi = 0; mi < 4; ++mi)
        if (do_mi[mi])
          acc[mi][ni] = wmma_bf16(af[mi].v, bf.v, acc[mi][ni]);
    }
    __syncthreads();
  }

  #pragma unroll
  for (int mi = 0; mi < 4; ++mi)
    #pragma unroll
    for (int ni = 0; ni < 4; ++ni)
      #pragma unroll
      for (int r = 0; r < 8; ++r) {
        int row = mb + wm * 64 + mi * 16 + 8 * half + r;
        int col = nb + wn * 64 + ni * 16 + ln;
        out[(size_t)row * DDIM + col] = acc[mi][ni][r];
      }
}

extern "C" void kernel_launch(void* const* d_in, const int* in_sizes, int n_in,
                              void* d_out, int out_size, void* d_ws, size_t ws_size,
                              hipStream_t stream) {
  (void)in_sizes; (void)n_in; (void)out_size; (void)ws_size;
  const float* x  = (const float*)d_in[0];
  const float* Wq = (const float*)d_in[1];
  const float* Wk = (const float*)d_in[2];
  const float* Wv = (const float*)d_in[3];
  float* out = (float*)d_out;

  unsigned short* xb = (unsigned short*)d_ws;
  unsigned short* Wb = xb + (size_t)SEQn * DDIM;
  unsigned short* Qb = Wb + (size_t)3 * DDIM * DDIM;
  unsigned short* Kb = Qb + (size_t)SEQn * DDIM;
  unsigned short* Vt = Kb + (size_t)SEQn * DDIM;
  float* rowM = (float*)(Vt + (size_t)DDIM * SEQn);
  float* rowL = rowM + SEQn;
  unsigned short* P = (unsigned short*)(rowL + SEQn);

  const int nx = SEQn * DDIM;
  const int nw = DDIM * DDIM;
  k_cvt_bf16<<<nx / (256 * 4), 256, 0, stream>>>(x,  xb,          nx);
  k_cvt_bf16<<<nw / (256 * 4), 256, 0, stream>>>(Wq, Wb,          nw);
  k_cvt_bf16<<<nw / (256 * 4), 256, 0, stream>>>(Wk, Wb + nw,     nw);
  k_cvt_bf16<<<nw / (256 * 4), 256, 0, stream>>>(Wv, Wb + 2 * nw, nw);

  k_qkv_gemm<<<dim3(DDIM / 256, SEQn / 128, 3), 256, 0, stream>>>(xb, Wb, Qb, Kb, Vt);

  k_softmax_stats<<<SEQn / 16, 128, 0, stream>>>(Qb, Kb, rowM, rowL);

  k_write_p<<<dim3(SEQn / 16, SEQn / 16), 32, 0, stream>>>(Qb, Kb, rowM, rowL, P);

  k_pv_gemm<<<dim3(DDIM / 256, SEQn / 128), 256, 0, stream>>>(P, Vt, out);
}
